// ALSHConv_26645977104459
// MI455X (gfx1250) — compile-verified, matrix operand order checked by reading, not run on hardware
//
#include <hip/hip_runtime.h>
#include <stdint.h>

// ---------------- problem constants ----------------
#define TABLE_SZ   1024
#define R_W        2.5f
#define U_B        0.83f
#define NPOW       9            // appended ALSH dims
#define C_IN       128
#define O_OUT      256
#define NIMG       16
#define P_IMG      4096         // 64*64
#define KK         1152         // C_IN * 9
#define M_TOT      65536        // NIMG * P_IMG
#define OUT_ELEMS  16777216     // NIMG * O_OUT * P_IMG

// ---------------- GEMM tiling ----------------
#define BM   128
#define BN   64
#define BK   32
#define LDA  40                 // padded LDS pitch (elements); 80B rows (16B multiple)
#define LDB  40                 // B pitch: 64B data + 16B pad  -> TDM pad 4 DW per 16 DW

#if defined(__has_builtin)
#  if __has_builtin(__builtin_amdgcn_tensor_load_to_lds)
#    define HAVE_TDM 1
#  endif
#endif

typedef __attribute__((ext_vector_type(16))) __bf16 v16bf;
typedef __attribute__((ext_vector_type(8)))  float  v8f;
typedef unsigned int u32x4 __attribute__((ext_vector_type(4)));
typedef int          i32x4 __attribute__((ext_vector_type(4)));
typedef int          i32x8 __attribute__((ext_vector_type(8)));

static __device__ __forceinline__ unsigned short f2bf(float f) {
    union { float f; uint32_t u; } cv; cv.f = f;
    uint32_t u = cv.u;
    u += 0x7FFFu + ((u >> 16) & 1u);      // round-to-nearest-even
    return (unsigned short)(u >> 16);
}

// ---- Stage 1: per-row ||k||^2 and k·a[:1152] ----
__global__ void k_row_stats(const float* __restrict__ kern, const float* __restrict__ a,
                            float* __restrict__ norms2, float* __restrict__ dots) {
    __shared__ float s1[256], s2[256];
    const int o = blockIdx.x, t = threadIdx.x;
    const float* row = kern + o * KK;
    float ss = 0.f, dd = 0.f;
    for (int k = t; k < KK; k += 256) { float v = row[k]; ss += v * v; dd += v * a[k]; }
    s1[t] = ss; s2[t] = dd; __syncthreads();
    for (int w = 128; w > 0; w >>= 1) {
        if (t < w) { s1[t] += s1[t + w]; s2[t] += s2[t + w]; }
        __syncthreads();
    }
    if (t == 0) { norms2[o] = s1[0]; dots[o] = s2[0]; }
}

// ---- Stage 2: scale = U/max||k||, hash each row -> bucket[o]; also zero count[] ----
__global__ void k_hash_table(const float* __restrict__ norms2, const float* __restrict__ dots,
                             const float* __restrict__ a, const float* __restrict__ bptr,
                             int* __restrict__ bucket, int* __restrict__ count) {
    const int t = threadIdx.x;
    for (int i = t; i < TABLE_SZ; i += 256) count[i] = 0;
    __shared__ float smax[256];
    smax[t] = norms2[t];
    __syncthreads();
    for (int w = 128; w > 0; w >>= 1) {
        if (t < w) smax[t] = fmaxf(smax[t], smax[t + w]);
        __syncthreads();
    }
    const float scale = U_B / sqrtf(smax[0]);
    float acc = scale * dots[t];
    float p = scale * scale * norms2[t];            // ||x||^2
    #pragma unroll
    for (int i = 0; i < NPOW; i++) { acc += p * a[KK + i]; p = p * p; }
    const float h = floorf((acc + bptr[0]) / R_W);
    int m = ((int)h) % TABLE_SZ;
    bucket[t] = m < 0 ? -m : m;
}

// ---- Stage 3: vote conv (GEMV) + histogram ----
__global__ void k_vote(const float* __restrict__ in, const float* __restrict__ a,
                       const float* __restrict__ bptr, int* __restrict__ count) {
    const int pix = blockIdx.x * 256 + threadIdx.x;      // 65536 total
    const int b = pix >> 12, p = pix & 4095, y = p >> 6, x = p & 63;
    const float* base = in + (size_t)b * (C_IN * P_IMG);
    float acc = 0.f;
    for (int c = 0; c < C_IN; ++c) {
        const float* ch = base + c * P_IMG;
        const float* ak = a + c * 9;
        #pragma unroll
        for (int ky = 0; ky < 3; ky++) {
            const int iy = y + ky - 1;
            if ((unsigned)iy >= 64u) continue;
            #pragma unroll
            for (int kx = 0; kx < 3; kx++) {
                const int ix = x + kx - 1;
                if ((unsigned)ix >= 64u) continue;
                acc += ch[iy * 64 + ix] * ak[ky * 3 + kx];
            }
        }
    }
    // constant-0.5 extra channel (zero padded at borders)
    const float* a9 = a + KK;
    #pragma unroll
    for (int ky = 0; ky < 3; ky++) {
        const int iy = y + ky - 1;
        if ((unsigned)iy >= 64u) continue;
        #pragma unroll
        for (int kx = 0; kx < 3; kx++) {
            const int ix = x + kx - 1;
            if ((unsigned)ix >= 64u) continue;
            acc += 0.5f * a9[ky * 3 + kx];
        }
    }
    int v = (int)floorf((acc + bptr[0]) / R_W);
    int m = v % TABLE_SZ; if (m < 0) m = -m;
    atomicAdd(&count[m], 1);
}

// ---- Stage 4: argmax over count (first-index tie break) ----
__global__ void k_argmax(const int* __restrict__ count, int* __restrict__ widx,
                         float* __restrict__ out_idx) {
    __shared__ int sc[256], si[256];
    const int t = threadIdx.x;
    int bc = -1, bi = 0;
    for (int i = t; i < TABLE_SZ; i += 256) {
        const int c = count[i];
        if (c > bc) { bc = c; bi = i; }
    }
    sc[t] = bc; si[t] = bi; __syncthreads();
    for (int w = 128; w > 0; w >>= 1) {
        if (t < w) {
            if (sc[t + w] > sc[t] || (sc[t + w] == sc[t] && si[t + w] < si[t])) {
                sc[t] = sc[t + w]; si[t] = si[t + w];
            }
        }
        __syncthreads();
    }
    if (t == 0) { widx[0] = si[0]; out_idx[0] = (float)si[0]; }
}

// ---- Stage 5: mask + masked weights -> bf16 ----
__global__ void k_mask_wbf(const float* __restrict__ kern, const int* __restrict__ bucket,
                           const int* __restrict__ widx, unsigned short* __restrict__ wbf,
                           float* __restrict__ out_mask) {
    const int o = blockIdx.x, t = threadIdx.x;
    const int msk = (bucket[o] == widx[0]) ? 1 : 0;
    if (t == 0) out_mask[o] = (float)msk;
    const float* row = kern + o * KK;
    unsigned short* wrow = wbf + o * KK;
    for (int k = t; k < KK; k += 256)
        wrow[k] = msk ? f2bf(row[k]) : (unsigned short)0;
}

// ---- Stage 6: input f32 -> bf16 (vectorized) ----
__global__ void k_in_bf(const float* __restrict__ in, unsigned short* __restrict__ inbf) {
    const int i = (blockIdx.x * 256 + threadIdx.x) * 4;
    const float4 v = *(const float4*)(in + i);
    ushort4 r;
    r.x = f2bf(v.x); r.y = f2bf(v.y); r.z = f2bf(v.z); r.w = f2bf(v.w);
    *(ushort4*)(inbf + i) = r;
}

// ---- Stage 7: implicit-GEMM conv via bf16 WMMA (f32 acc) + TDM weight staging ----
// M = 65536 (b,y,x), N = 256 (o), K = 1152 (c,ky,kx)
__launch_bounds__(256)
__global__ void k_conv_wmma(const unsigned short* __restrict__ inbf,
                            const unsigned short* __restrict__ wbf,
                            float* __restrict__ out) {
    __shared__ __attribute__((aligned(16))) unsigned short As[BM * LDA]; // 10.0 KB
    __shared__ __attribute__((aligned(16))) unsigned short Bs[BN * LDB]; //  5.0 KB

    const int tid    = threadIdx.x;
    const int m_base = blockIdx.x * BM;       // 0..65408
    const int n_base = blockIdx.y * BN;       // 0..192
    const int img    = m_base >> 12;          // whole BM tile stays in one image
    const int p_base = m_base & 4095;
    const unsigned short* in_img = inbf + (size_t)img * (C_IN * P_IMG);

    const int wid   = tid >> 5;               // 8 waves: 4 along M, 2 along N
    const int lane  = tid & 31;
    const int wm    = (wid & 3) * 32;
    const int wn    = (wid >> 2) * 32;
    const int lhalf = lane >> 4;              // lane half selects K sub-block
    const int lrow  = lane & 15;              // row (A: M, B: N) within 16

    // hoisted per-thread A-gather geometry (i, t0 fixed per thread)
    const int ga_i  = tid & 127;              // m within tile
    const int ga_t0 = tid >> 7;               // k phase (0/1), k = kb + t0 + 2*it
    const int ga_p  = p_base + ga_i;
    const int ga_y  = ga_p >> 6;
    const int ga_x  = ga_p & 63;

#ifdef HAVE_TDM
    // --- Tensor DMA descriptor (D#) for the 64x32 bf16 weight tile, LDS-padded ---
    // group0: count=1 | lds_addr | global_addr | type=2
    const uint32_t ldsB  = (uint32_t)(uintptr_t)(&Bs[0]);
    const uint64_t gbase = (uint64_t)(uintptr_t)wbf + 2ull * ((uint64_t)n_base * KK);
    // group1 (constant): data_size=1(2B), pad_enable, pad_interval=3 (16 DW = 64B row),
    //                    pad_amount=3 (4 DW = 16B) -> LDS pitch 80B == LDB
    i32x8 g1;
    g1[0] = (1 << 16) | (1 << 20) | (3 << 22) | (3 << 25);
    g1[1] = (int)(((uint32_t)KK) << 16);          // tensor_dim0[15:0] = 1152
    g1[2] = (int)(((uint32_t)BN) << 16);          // tensor_dim0 hi=0 | tensor_dim1 = 64
    g1[3] = (int)(((uint32_t)BK) << 16);          // tensor_dim1 hi=0 | tile_dim0 = 32
    g1[4] = BN;                                   // tile_dim1 = 64, tile_dim2 = 0
    g1[5] = KK;                                   // tensor_dim0_stride = 1152
    g1[6] = 0;
    g1[7] = 0;
    const i32x4 gz = {0, 0, 0, 0};                // groups 2/3 unused (2D tensor)
#if !defined(__clang_major__) || (__clang_major__ >= 23)
    const i32x8 gz8 = {0, 0, 0, 0, 0, 0, 0, 0};
#endif
#endif

    const v8f vzero = {};
    v8f acc[2][2];
    #pragma unroll
    for (int i = 0; i < 2; i++)
        #pragma unroll
        for (int j = 0; j < 2; j++) acc[i][j] = vzero;

    for (int kb = 0; kb < KK; kb += BK) {
        // prefetch next weight K-slab into L2 (global_prefetch path)
        if (kb + BK < KK)
            __builtin_prefetch(wbf + (size_t)n_base * KK + kb + BK, 0, 1);

#ifdef HAVE_TDM
        // -- B tile via Tensor Data Mover: global (stride 1152) -> LDS (pitch 40) --
        if (wid == 0) {
            const uint64_t ga = gbase + 2ull * (uint64_t)kb;
            u32x4 g0;
            g0.x = 1u;                                             // count=1
            g0.y = ldsB;                                           // lds_addr
            g0.z = (uint32_t)ga;                                   // global_addr[31:0]
            g0.w = ((uint32_t)(ga >> 32) & 0x1FFFFFFu) | (2u << 30); // addr[56:32] | type=2
#if defined(__clang_major__) && (__clang_major__ < 23)
            __builtin_amdgcn_tensor_load_to_lds(g0, g1, gz, gz, 0);          // ROCm 7.2 (5-arg)
#else
            __builtin_amdgcn_tensor_load_to_lds(g0, g1, gz, gz, gz8, 0);     // clang-23 (6-arg)
#endif
            __builtin_amdgcn_s_wait_tensorcnt(0);
        }
#else
        // -- fallback: B tile via vector loads, stored N-major --
        #pragma unroll
        for (int it = 0; it < 8; ++it) {
            const int e = it * 256 + tid;
            const int t = e & 31;
            const int j = e >> 5;
            Bs[j * LDB + t] = wbf[(size_t)(n_base + j) * KK + kb + t];
        }
#endif

        // -- A tile: 128 x 32, im2col gather of bf16 input, coalesced along M --
        for (int it = 0; it < 16; ++it) {
            const int t  = ga_t0 + 2 * it;    // k offset within block
            const int k  = kb + t;
            const int c  = k / 9;
            const int rr = k - c * 9;
            const int ky = rr / 3;
            const int kx = rr - ky * 3;
            const int iy = ga_y + ky - 1;
            const int ix = ga_x + kx - 1;
            unsigned short v = 0;
            if ((unsigned)iy < 64u && (unsigned)ix < 64u)
                v = in_img[c * P_IMG + iy * 64 + ix];
            As[ga_i * LDA + t] = v;
        }
        __syncthreads();

        // -- fragments per ISA 16-bit layout: VGPR0-3 K=h*8.., VGPR4-7 K=16+h*8.. --
        union Frag { v16bf v; struct { uint4 lo; uint4 hi; } q; };
        Frag a0, a1, b0, b1;
        {
            const unsigned short* pa0 = &As[(wm + lrow) * LDA + lhalf * 8];
            a0.q.lo = *(const uint4*)(pa0);
            a0.q.hi = *(const uint4*)(pa0 + 16);
            const unsigned short* pa1 = &As[(wm + 16 + lrow) * LDA + lhalf * 8];
            a1.q.lo = *(const uint4*)(pa1);
            a1.q.hi = *(const uint4*)(pa1 + 16);
            const unsigned short* pb0 = &Bs[(wn + lrow) * LDB + lhalf * 8];
            b0.q.lo = *(const uint4*)(pb0);
            b0.q.hi = *(const uint4*)(pb0 + 16);
            const unsigned short* pb1 = &Bs[(wn + 16 + lrow) * LDB + lhalf * 8];
            b1.q.lo = *(const uint4*)(pb1);
            b1.q.hi = *(const uint4*)(pb1 + 16);
        }
        acc[0][0] = __builtin_amdgcn_wmma_f32_16x16x32_bf16(false, a0.v, false, b0.v,
                                                            (short)0, acc[0][0], false, false);
        acc[0][1] = __builtin_amdgcn_wmma_f32_16x16x32_bf16(false, a0.v, false, b1.v,
                                                            (short)0, acc[0][1], false, false);
        acc[1][0] = __builtin_amdgcn_wmma_f32_16x16x32_bf16(false, a1.v, false, b0.v,
                                                            (short)0, acc[1][0], false, false);
        acc[1][1] = __builtin_amdgcn_wmma_f32_16x16x32_bf16(false, a1.v, false, b1.v,
                                                            (short)0, acc[1][1], false, false);
        __syncthreads();
    }

    // -- epilogue: per lane, 8 consecutive M at fixed N -> two 16B stores --
    #pragma unroll
    for (int ti = 0; ti < 2; ++ti) {
        const int mloc = wm + ti * 16 + lhalf * 8;
        #pragma unroll
        for (int tj = 0; tj < 2; ++tj) {
            const int n = n_base + wn + tj * 16 + lrow;
            float* dst = out + (((size_t)(img * O_OUT + n)) << 12) + p_base + mloc;
            const float* av = (const float*)&acc[ti][tj];
            *(float4*)(dst)     = *(const float4*)(av);
            *(float4*)(dst + 4) = *(const float4*)(av + 4);
        }
    }
}

extern "C" void kernel_launch(void* const* d_in, const int* in_sizes, int n_in,
                              void* d_out, int out_size, void* d_ws, size_t ws_size,
                              hipStream_t stream) {
    const float* in   = (const float*)d_in[0];  // [16,128,64,64]
    const float* kern = (const float*)d_in[1];  // [256,128,3,3]
    const float* a    = (const float*)d_in[2];  // [1161]
    const float* b    = (const float*)d_in[3];  // [1]
    float* out = (float*)d_out;                 // conv out | index | mask (flattened)

    char* ws = (char*)d_ws;
    float* norms2          = (float*)(ws);                       //  1 KB
    float* dots            = (float*)(ws + 1024);                //  1 KB
    int*   bucket          = (int*)  (ws + 2048);                //  1 KB
    int*   count           = (int*)  (ws + 3072);                //  4 KB
    int*   widx            = (int*)  (ws + 7168);                //  4 B
    unsigned short* wbf    = (unsigned short*)(ws + 8192);       //  256*1152*2 = 576 KB
    unsigned short* inbf   = (unsigned short*)(ws + 8192 + 589824); // 16 MB bf16 input

    k_row_stats <<<O_OUT, 256, 0, stream>>>(kern, a, norms2, dots);
    k_hash_table<<<1,     256, 0, stream>>>(norms2, dots, a, b, bucket, count);
    k_vote      <<<M_TOT / 256, 256, 0, stream>>>(in, a, b, count);
    k_argmax    <<<1,     256, 0, stream>>>(count, widx, out + OUT_ELEMS);
    k_mask_wbf  <<<O_OUT, 256, 0, stream>>>(kern, bucket, widx, wbf, out + OUT_ELEMS + 1);
    k_in_bf     <<<(NIMG * C_IN * P_IMG) / (256 * 4), 256, 0, stream>>>(in, inbf);

    dim3 grid(M_TOT / BM, O_OUT / BN);          // 512 x 4
    k_conv_wmma <<<grid, 256, 0, stream>>>(inbf, wbf, out);
}
